// CacaAttention_87917980549743
// MI455X (gfx1250) — compile-verified, matrix operand order checked by reading, not run on hardware
//
#include <hip/hip_runtime.h>
#include <hip/hip_bf16.h>
#include <stdint.h>

typedef __bf16 bhalf;
typedef __attribute__((ext_vector_type(16))) __bf16 v16bf;
typedef __attribute__((ext_vector_type(8)))  float  v8f;

#define BATCH 2
#define SEQ   2048
#define HID   2048
#define NH    16
#define NKV   4
#define HD    128

// ---------------------------------------------------------------------------
// Optional CDNA5 async global->LDS path (ASYNCcnt-tracked), with safe fallback.
// Builtin expects: (v4i addrspace(1)*, v4i addrspace(3)*, imm offset, imm cpol)
// ---------------------------------------------------------------------------
#if __has_builtin(__builtin_amdgcn_global_load_async_to_lds_b128) && \
    __has_builtin(__builtin_amdgcn_s_wait_asynccnt)
#define HAVE_ASYNC_LDS 1
#else
#define HAVE_ASYNC_LDS 0
#endif

typedef int v4i_g __attribute__((vector_size(16)));
typedef __attribute__((address_space(1))) v4i_g as1_v4i;
typedef __attribute__((address_space(3))) v4i_g as3_v4i;

__device__ __forceinline__ void copy16_to_lds(const bhalf* g, bhalf* l) {
#if HAVE_ASYNC_LDS
  // integer->pointer casts avoid addrspacecast legality issues; low 32 bits of
  // a generic shared pointer are the LDS byte offset (aperture is in hi bits).
  __builtin_amdgcn_global_load_async_to_lds_b128(
      (as1_v4i*)(uintptr_t)g,
      (as3_v4i*)(unsigned)(uintptr_t)l, 0, 0);
#else
  *(uint4*)l = *(const uint4*)g;
#endif
}

__device__ __forceinline__ void async_lds_fence() {
#if HAVE_ASYNC_LDS
  __builtin_amdgcn_s_wait_asynccnt(0);
#endif
}

// ---------------------------------------------------------------------------
// CDNA5 WMMA fragment helpers (wave32, 16x16x32 bf16)
// A-matrix 16x32 per-lane layout (ISA 7.12.2): lane = {m = lane&15, half = lane>>4}
//   vgpr v in 0..3 : K = half*8 + 2v + {0,1}
//   vgpr v in 4..7 : K = 16 + half*8 + 2(v-4) + {0,1}
// B-matrix 32x16: lane holds column n = lane&15, same K mapping.
// ---------------------------------------------------------------------------
__device__ __forceinline__ int frag_k(int v, int h) {
  return ((v & 4) ? 16 : 0) + ((v & 3) << 1) + (h << 3);
}

// Row-major source: lane picks row (lane&15), walks K along the row.
// Per lane this is two contiguous 16B runs -> vectorizes to b128 loads.
__device__ __forceinline__ v16bf load_frag_row(const bhalf* base, int ld, int lane) {
  const int m = lane & 15, h = lane >> 4;
  const bhalf* rp = base + (size_t)m * ld;
  v16bf f;
#pragma unroll
  for (int v = 0; v < 8; ++v) {
    const int k = frag_k(v, h);
    f[2 * v]     = rp[k];
    f[2 * v + 1] = rp[k + 1];
  }
  return f;
}

// Row-major KxN source read as B-matrix: lane picks column (lane&15), strides rows.
__device__ __forceinline__ v16bf load_frag_col(const bhalf* base, int ld, int lane) {
  const int n = lane & 15, h = lane >> 4;
  v16bf f;
#pragma unroll
  for (int v = 0; v < 8; ++v) {
    const int k = frag_k(v, h);
    f[2 * v]     = base[(size_t)k * ld + n];
    f[2 * v + 1] = base[(size_t)(k + 1) * ld + n];
  }
  return f;
}

__device__ __forceinline__ v8f wmma_bf16(v16bf a, v16bf b, v8f c) {
  return __builtin_amdgcn_wmma_f32_16x16x32_bf16(false, a, false, b,
                                                 (short)0, c, false, false);
}

// ---------------------------------------------------------------------------
// fp32 -> bf16 cast
// ---------------------------------------------------------------------------
__global__ void cast_f32_bf16(const float* __restrict__ in,
                              bhalf* __restrict__ out, long n) {
  long i = (long)blockIdx.x * blockDim.x + threadIdx.x;
  const long stride = (long)gridDim.x * blockDim.x;
  for (; i < n; i += stride) out[i] = (bhalf)in[i];
}

// ---------------------------------------------------------------------------
// Tiled bf16 GEMM: C[M,N] = A[M,K] * B[K,N], all row-major.
// 256 threads = 8 waves arranged 4(M) x 2(N); block tile 128x128, BK = 32.
// Wave tile 32x64: 2 A-frags x 4 B-frags -> 8 WMMAs per K-step, B reused 2x.
// Staging uses GLOBAL_LOAD_ASYNC_TO_LDS_B128 when available.
// ---------------------------------------------------------------------------
#define BM 128
#define BN 128
#define BK 32
#define GTHREADS 256
#define LDA_S (BK + 8)   // 40 elems = 80B rows  (16B aligned chunks)
#define LDB_S (BN + 8)   // 136 elems = 272B rows (16B aligned chunks)

template <bool OUT_F32>
__global__ __launch_bounds__(GTHREADS) void gemm_bf16_wmma(
    const bhalf* __restrict__ A, const bhalf* __restrict__ Bm,
    void* __restrict__ Cout, int M, int N, int K) {
  __shared__ bhalf lA[BM * LDA_S];
  __shared__ bhalf lB[BK * LDB_S];
  const int tid  = threadIdx.x;
  const int lane = tid & 31;
  const int w    = tid >> 5;
  const int wm   = w & 3;        // wave row 0..3  -> rows wm*32..wm*32+31
  const int wn   = w >> 2;       // wave col 0..1  -> cols wn*64..wn*64+63
  const int m0   = blockIdx.y * BM;
  const int n0   = blockIdx.x * BN;

  const v8f vzero = {0.f, 0.f, 0.f, 0.f, 0.f, 0.f, 0.f, 0.f};
  v8f acc[8];
#pragma unroll
  for (int c = 0; c < 8; ++c) acc[c] = vzero;

  for (int k0 = 0; k0 < K; k0 += BK) {
    // Stage A (128x32) and B (32x128) tiles, 16B per transfer.
#pragma unroll
    for (int i = tid * 8; i < BM * BK; i += GTHREADS * 8) {
      const int r = i / BK, c = i % BK;
      copy16_to_lds(&A[(size_t)(m0 + r) * K + k0 + c], &lA[r * LDA_S + c]);
    }
#pragma unroll
    for (int i = tid * 8; i < BK * BN; i += GTHREADS * 8) {
      const int r = i / BN, c = i % BN;
      copy16_to_lds(&Bm[(size_t)(k0 + r) * N + n0 + c], &lB[r * LDB_S + c]);
    }
    if (k0 + BK < K) {  // global_prefetch_b8 for the next K tile
      __builtin_prefetch(&A[(size_t)(m0 + (tid >> 1)) * K + k0 + BK], 0, 1);
      __builtin_prefetch(&Bm[(size_t)(k0 + BK + (tid >> 3)) * N + n0], 0, 1);
    }
    async_lds_fence();
    __syncthreads();

    const v16bf af0 = load_frag_row(&lA[(wm * 32) * LDA_S], LDA_S, lane);
    const v16bf af1 = load_frag_row(&lA[(wm * 32 + 16) * LDA_S], LDA_S, lane);
#pragma unroll
    for (int c = 0; c < 4; ++c) {
      const v16bf bf = load_frag_col(&lB[wn * 64 + c * 16], LDB_S, lane);
      acc[c]     = wmma_bf16(af0, bf, acc[c]);
      acc[4 + c] = wmma_bf16(af1, bf, acc[4 + c]);
    }
    __syncthreads();
  }

  const int n = lane & 15, hh = lane >> 4;
#pragma unroll
  for (int s = 0; s < 2; ++s) {
#pragma unroll
    for (int c = 0; c < 4; ++c) {
#pragma unroll
      for (int r = 0; r < 8; ++r) {
        const int row = m0 + wm * 32 + s * 16 + r + hh * 8;
        const int col = n0 + wn * 64 + c * 16 + n;
        if (OUT_F32)
          ((float*)Cout)[(size_t)row * N + col] = acc[s * 4 + c][r];
        else
          ((bhalf*)Cout)[(size_t)row * N + col] = (bhalf)acc[s * 4 + c][r];
      }
    }
  }
}

// ---------------------------------------------------------------------------
// RoPE in place on bf16 buffer laid out [B*S, nheads*128].
// ---------------------------------------------------------------------------
__global__ void rope_kernel(bhalf* __restrict__ buf, int nheads) {
  const int idx = blockIdx.x * blockDim.x + threadIdx.x;
  const int total = BATCH * SEQ * nheads * (HD / 2);
  if (idx >= total) return;
  const int i = idx & 63;
  const int h = (idx >> 6) % nheads;
  const int s = (idx / (64 * nheads)) % SEQ;
  const int b = idx / (64 * nheads * SEQ);
  const size_t row = ((size_t)(b * SEQ + s) * nheads + h) * HD;
  const float invf = __powf(10000.0f, -2.0f * (float)i / 128.0f);
  const float ang  = (float)s * invf;
  const float c = __cosf(ang), sn = __sinf(ang);
  const float x1 = (float)buf[row + i];
  const float x2 = (float)buf[row + 64 + i];
  buf[row + i]      = (bhalf)(x1 * c - x2 * sn);
  buf[row + 64 + i] = (bhalf)(x2 * c + x1 * sn);
}

// ---------------------------------------------------------------------------
// Causal flash attention with GQA. One wave per 16-query block of one head.
// Q: [B*S, 2048] bf16 (post-RoPE), K/V: [B*S, 512] bf16, Ctx out: [B*S, 2048].
// 32 keys/iteration: 8 WMMAs QK^T, online softmax (16-lane butterflies),
// P restaged via per-wave LDS, 8 WMMAs PV.
// ---------------------------------------------------------------------------
__global__ __launch_bounds__(128) void flash_attn_wmma(
    const bhalf* __restrict__ Q, const bhalf* __restrict__ Kc,
    const bhalf* __restrict__ Vc, bhalf* __restrict__ Ctx) {
  __shared__ bhalf ldsP[4][16 * 32];
  const int tid  = threadIdx.x;
  const int lane = tid & 31;
  const int w    = tid >> 5;
  const int task = blockIdx.x * 4 + w;     // [0, B*NH*(SEQ/16))
  const int qblk = task & 127;
  const int h    = (task >> 7) & 15;
  const int b    = task >> 11;
  const int kv   = h >> 2;                 // GQA: 4 q-heads per kv-head
  const int q0   = qblk * 16;
  const float scale = 0.08838834764831845f;  // 1/sqrt(128)
  const int hh = lane >> 4;
  const int ln = lane & 15;
  const size_t kvld = (size_t)NKV * HD;    // 512

  const bhalf* qbase = Q + ((size_t)(b * SEQ) + q0) * (NH * HD) + h * HD;
  v16bf qa[4];
#pragma unroll
  for (int dc = 0; dc < 4; ++dc)
    qa[dc] = load_frag_row(qbase + dc * 32, NH * HD, lane);

  const v8f vzero = {0.f, 0.f, 0.f, 0.f, 0.f, 0.f, 0.f, 0.f};
  v8f acc[8];
#pragma unroll
  for (int t = 0; t < 8; ++t) acc[t] = vzero;
  float mrow[8], lrow[8];
#pragma unroll
  for (int r = 0; r < 8; ++r) { mrow[r] = -1e30f; lrow[r] = 0.0f; }

  const int kend = q0 + 16;  // causal upper bound (exclusive)
  for (int j = 0; j < kend; j += 32) {
    const bhalf* kbase = Kc + ((size_t)(b * SEQ) + j) * kvld + kv * HD;
    v8f s0 = vzero, s1 = vzero;
#pragma unroll
    for (int dc = 0; dc < 4; ++dc) {
      const v16bf kb0 = load_frag_row(kbase + dc * 32, (int)kvld, lane);
      const v16bf kb1 = load_frag_row(kbase + 16 * kvld + dc * 32, (int)kvld, lane);
      s0 = wmma_bf16(qa[dc], kb0, s0);
      s1 = wmma_bf16(qa[dc], kb1, s1);
    }

    float p0[8], p1[8], tmax[8];
#pragma unroll
    for (int r = 0; r < 8; ++r) {
      const int qrow = q0 + r + hh * 8;
      const float v0 = (j + ln <= qrow)      ? s0[r] * scale : -1e30f;
      const float v1 = (j + 16 + ln <= qrow) ? s1[r] * scale : -1e30f;
      p0[r] = v0;
      p1[r] = v1;
      tmax[r] = fmaxf(v0, v1);
    }
#pragma unroll
    for (int d = 1; d < 16; d <<= 1)
#pragma unroll
      for (int r = 0; r < 8; ++r)
        tmax[r] = fmaxf(tmax[r], __shfl_xor(tmax[r], d, 32));

    float fac[8], rsum[8];
#pragma unroll
    for (int r = 0; r < 8; ++r) {
      const float mnew = fmaxf(mrow[r], tmax[r]);
      fac[r] = __expf(mrow[r] - mnew);
      p0[r]  = __expf(p0[r] - mnew);
      p1[r]  = __expf(p1[r] - mnew);
      rsum[r] = p0[r] + p1[r];
      mrow[r] = mnew;
    }
#pragma unroll
    for (int d = 1; d < 16; d <<= 1)
#pragma unroll
      for (int r = 0; r < 8; ++r)
        rsum[r] += __shfl_xor(rsum[r], d, 32);
#pragma unroll
    for (int r = 0; r < 8; ++r) lrow[r] = lrow[r] * fac[r] + rsum[r];
#pragma unroll
    for (int t = 0; t < 8; ++t)
#pragma unroll
      for (int r = 0; r < 8; ++r) acc[t][r] *= fac[r];

    // accumulator layout -> A-fragment layout via per-wave LDS tile
#pragma unroll
    for (int r = 0; r < 8; ++r) {
      const int m = r + hh * 8;
      ldsP[w][m * 32 + ln]      = (bhalf)p0[r];
      ldsP[w][m * 32 + 16 + ln] = (bhalf)p1[r];
    }
    const v16bf pa = load_frag_row(ldsP[w], 32, lane);

    const bhalf* vbase = Vc + ((size_t)(b * SEQ) + j) * kvld + kv * HD;
#pragma unroll
    for (int t = 0; t < 8; ++t) {
      const v16bf vb = load_frag_col(vbase + t * 16, (int)kvld, lane);
      acc[t] = wmma_bf16(pa, vb, acc[t]);
    }
  }

  bhalf* obase = Ctx + ((size_t)(b * SEQ) + q0) * (NH * HD) + h * HD;
#pragma unroll
  for (int t = 0; t < 8; ++t)
#pragma unroll
    for (int r = 0; r < 8; ++r) {
      const int m = r + hh * 8;
      obase[(size_t)m * (NH * HD) + t * 16 + ln] = (bhalf)(acc[t][r] / lrow[r]);
    }
}

// ---------------------------------------------------------------------------
// Launch
// ---------------------------------------------------------------------------
extern "C" void kernel_launch(void* const* d_in, const int* in_sizes, int n_in,
                              void* d_out, int out_size, void* d_ws, size_t ws_size,
                              hipStream_t stream) {
  (void)in_sizes; (void)n_in; (void)out_size; (void)ws_size;
  const float* X  = (const float*)d_in[0];
  const float* Wq = (const float*)d_in[1];
  const float* Wk = (const float*)d_in[2];
  const float* Wv = (const float*)d_in[3];
  const float* Wo = (const float*)d_in[4];

  const size_t nX  = (size_t)BATCH * SEQ * HID;   // 8M
  const size_t nWq = (size_t)HID * HID;           // 4M
  const size_t nWk = (size_t)HID * (NKV * HD);    // 1M
  const size_t nKV = (size_t)BATCH * SEQ * (NKV * HD);

  bhalf* ws  = (bhalf*)d_ws;
  bhalf* Xb  = ws;            ws += nX;
  bhalf* Wqb = ws;            ws += nWq;
  bhalf* Wkb = ws;            ws += nWk;
  bhalf* Wvb = ws;            ws += nWk;
  bhalf* Wob = ws;            ws += nWq;
  bhalf* Qb  = ws;            ws += nX;
  bhalf* Kb  = ws;            ws += nKV;
  bhalf* Vb  = ws;            ws += nKV;
  bhalf* Ctx = ws;            ws += nX;

  // 1) fp32 -> bf16
  cast_f32_bf16<<<4096, 256, 0, stream>>>(X,  Xb,  (long)nX);
  cast_f32_bf16<<<4096, 256, 0, stream>>>(Wq, Wqb, (long)nWq);
  cast_f32_bf16<<<2048, 256, 0, stream>>>(Wk, Wkb, (long)nWk);
  cast_f32_bf16<<<2048, 256, 0, stream>>>(Wv, Wvb, (long)nWk);
  cast_f32_bf16<<<4096, 256, 0, stream>>>(Wo, Wob, (long)nWq);

  const int M = BATCH * SEQ;  // 4096
  // 2) projections
  gemm_bf16_wmma<false><<<dim3(HID / BN, M / BM), GTHREADS, 0, stream>>>(
      Xb, Wqb, Qb, M, HID, HID);
  gemm_bf16_wmma<false><<<dim3((NKV * HD) / BN, M / BM), GTHREADS, 0, stream>>>(
      Xb, Wkb, Kb, M, NKV * HD, HID);
  gemm_bf16_wmma<false><<<dim3((NKV * HD) / BN, M / BM), GTHREADS, 0, stream>>>(
      Xb, Wvb, Vb, M, NKV * HD, HID);

  // 3) RoPE on Q and K
  rope_kernel<<<(BATCH * SEQ * NH * 64 + 255) / 256, 256, 0, stream>>>(Qb, NH);
  rope_kernel<<<(BATCH * SEQ * NKV * 64 + 255) / 256, 256, 0, stream>>>(Kb, NKV);

  // 4) causal flash attention: B*NH*(SEQ/16) wave-tasks, 4 waves/block
  flash_attn_wmma<<<(BATCH * NH * (SEQ / 16)) / 4, 128, 0, stream>>>(Qb, Kb, Vb, Ctx);

  // 5) output projection -> fp32
  gemm_bf16_wmma<true><<<dim3(HID / BN, M / BM), GTHREADS, 0, stream>>>(
      Ctx, Wob, (float*)d_out, M, HID, HID);
}
